// STGAT_46351287059149
// MI455X (gfx1250) — matrix operand order repaired and checked by reading, not compiled, not run on hardware
//
#include <hip/hip_runtime.h>

typedef __attribute__((ext_vector_type(16))) _Float16 v16h;
typedef __attribute__((ext_vector_type(8)))  _Float16 v8h;
typedef __attribute__((ext_vector_type(8)))  float    v8f;

__device__ __forceinline__ float fast_rcp(float x)  { return __builtin_amdgcn_rcpf(x); }
__device__ __forceinline__ float fast_rsq(float x)  { return __builtin_amdgcn_rsqf(x); }

// ---------------------------------------------------------------------------
// WMMA fragment loaders (CDNA5 16x16x32 f16 layouts, CDNA5 ISA)
// A 16x32 (MxK): lane<16 -> M=lane, halves=K(k0..k0+15) (contiguous k, same order as B)
//                lane>=16 -> same M, halves=K(k0+16..k0+31)
// B 32x16 (KxN) from Bt[N][K] row-major: lane<16 -> N=lane, halves=K(k0..k0+15)
//                lane>=16 -> N=lane-16, halves=K(k0+16..k0+31)  (one 32B load)
// C 16x16 f32: VGPR r -> M = r + (lane>=16 ? 8 : 0), N = lane&15
// ---------------------------------------------------------------------------
__device__ __forceinline__ v16h load_afrag(const _Float16* __restrict__ A, long lda,
                                           int mbase, int k0, int lane) {
  int m  = mbase + (lane & 15);
  int kb = k0 + ((lane & 16) ? 16 : 0);
  const _Float16* p = A + (long)m * lda + kb;
  v8h lo = *(const v8h*)p;
  v8h hi = *(const v8h*)(p + 8);
  v16h a;
#pragma unroll
  for (int i = 0; i < 8; ++i) { a[i] = lo[i]; a[i + 8] = hi[i]; }
  return a;
}

__device__ __forceinline__ v16h load_bfrag(const _Float16* __restrict__ Bt, long ldb,
                                           int nbase, int k0, int lane) {
  int n  = nbase + (lane & 15);
  int kb = k0 + ((lane & 16) ? 16 : 0);
  return *(const v16h*)(Bt + (long)n * ldb + kb);
}

// ---------------------------------------------------------------------------
// Generic batched GEMM; A and B fragments double-buffered across k-steps so
// current WMMAs consume resident registers while next-step loads are in
// flight (distinct regs -> no WAR hazard nops, no loadcnt-0 stalls).
// C[z] = A * Bt[z]^T ; A f16 [M,Kpad], Bt f16 [Nc,Kpad], C f32 [M,Nc].
// 4 waves/block, each wave -> 16(M) x NSUB*16(N) strip.
// ---------------------------------------------------------------------------
template <int NSUB>
__global__ void k_gemm_t(const _Float16* __restrict__ A, const _Float16* __restrict__ Bt,
                         float* __restrict__ C, int M, int Kpad, int Nc,
                         long strideBt, long strideC) {
  int wave = threadIdx.x >> 5, lane = threadIdx.x & 31;
  int mbase = (blockIdx.x * 4 + wave) * 16;
  if (mbase >= M) return;                       // wave-uniform, never taken here
  const _Float16* Bth = Bt + (long)blockIdx.z * strideBt;
  float* Ch = C + (long)blockIdx.z * strideC;
  int nb0 = blockIdx.y * (NSUB * 16);
  v8f acc[NSUB];
#pragma unroll
  for (int s = 0; s < NSUB; ++s)
#pragma unroll
    for (int r = 0; r < 8; ++r) acc[s][r] = 0.f;

  v16h a_cur = load_afrag(A, Kpad, mbase, 0, lane);
  v16h b_cur[NSUB];
#pragma unroll
  for (int s = 0; s < NSUB; ++s) b_cur[s] = load_bfrag(Bth, Kpad, nb0 + s * 16, 0, lane);

  for (int k0 = 0; k0 < Kpad; k0 += 32) {
    int kn = (k0 + 32 < Kpad) ? (k0 + 32) : k0;     // branch-free tail
    v16h a_nxt = load_afrag(A, Kpad, mbase, kn, lane);
    v16h b_nxt[NSUB];
#pragma unroll
    for (int s = 0; s < NSUB; ++s) b_nxt[s] = load_bfrag(Bth, Kpad, nb0 + s * 16, kn, lane);
#pragma unroll
    for (int s = 0; s < NSUB; ++s)
      acc[s] = __builtin_amdgcn_wmma_f32_16x16x32_f16(false, a_cur, false, b_cur[s],
                                                      (short)0, acc[s], false, false);
    a_cur = a_nxt;
#pragma unroll
    for (int s = 0; s < NSUB; ++s) b_cur[s] = b_nxt[s];
  }
  int col = lane & 15, rofs = (lane & 16) ? 8 : 0;
#pragma unroll
  for (int s = 0; s < NSUB; ++s)
#pragma unroll
    for (int r = 0; r < 8; ++r)
      Ch[(long)(mbase + r + rofs) * Nc + nb0 + s * 16 + col] = acc[s][r];
}

// ---------------------------------------------------------------------------
// im2col over time axis (kernel (1,3), same padding), fp32 -> fp16, K padded.
// x: [BN, T, C] ; col: [BN*T, Kpad] with k = tap*C + c
// ---------------------------------------------------------------------------
__global__ void k_im2col(const float* __restrict__ x, _Float16* __restrict__ col,
                         int BN, int T, int C, int Kpad) {
  long i = (long)blockIdx.x * blockDim.x + threadIdx.x;
  long total = (long)BN * T * Kpad;
  if (i >= total) return;
  int k = (int)(i % Kpad);
  long row = i / Kpad;
  int t = (int)(row % T);
  long bn = row / T;
  float v = 0.f;
  if (k < 3 * C) {
    int tap = k / C, c = k % C;
    int ts = t + tap - 1;
    if (ts >= 0 && ts < T) v = x[(bn * T + ts) * C + c];
  }
  col[i] = (_Float16)v;
}

// Concatenated conv weights -> Bt [192, Kpad]; W* layout [64, cin, 1, 3]
__global__ void k_build_wcat(const float* __restrict__ W1, const float* __restrict__ W2,
                             const float* __restrict__ W3, _Float16* __restrict__ out,
                             int C, int Kpad) {
  int i = blockIdx.x * blockDim.x + threadIdx.x;
  int total = 192 * Kpad;
  if (i >= total) return;
  int k = i % Kpad, o = i / Kpad;
  float v = 0.f;
  if (k < 3 * C) {
    const float* Ws = (o < 64) ? W1 : ((o < 128) ? W2 : W3);
    int oc = o & 63, tap = k / C, c = k % C;
    v = Ws[(oc * C + c) * 3 + tap];
  }
  out[i] = (_Float16)v;
}

// conv epilogue: relu(c1+b1 + sigmoid(c2+b2) + c3+b3) -> t16 [BNT,64]
__global__ void k_conv_epi(const float* __restrict__ raw, const float* __restrict__ b1,
                           const float* __restrict__ b2, const float* __restrict__ b3,
                           _Float16* __restrict__ t16, long Mrows) {
  long i = (long)blockIdx.x * blockDim.x + threadIdx.x;
  long total = Mrows * 64;
  if (i >= total) return;
  int o = (int)(i & 63);
  long row = i >> 6;
  const float* r = raw + row * 192;
  float c1 = r[o] + b1[o], c2 = r[64 + o] + b2[o], c3 = r[128 + o] + b3[o];
  float v = c1 + fast_rcp(1.f + __expf(-c2)) + c3;
  t16[i] = (_Float16)fmaxf(v, 0.f);
}

// per-head weight transpose fp32 [K,64] -> fp16 Bt [64,K]
__global__ void k_transpose_bt(const float* __restrict__ Win, _Float16* __restrict__ out,
                               int K, long strideIn, long strideOut) {
  int h = blockIdx.z;
  long i = (long)blockIdx.x * blockDim.x + threadIdx.x;
  long total = (long)K * 64;
  if (i >= total) return;
  int k = (int)(i % K), n = (int)(i / K);
  out[(long)h * strideOut + (long)n * K + k] = (_Float16)Win[(long)h * strideIn + (long)k * 64 + n];
}

// f1/f2 = sf . a1/a2 + bias
__global__ void k_f1f2(const float* __restrict__ sf, const float* __restrict__ a1,
                       const float* __restrict__ a2, const float* __restrict__ ba1,
                       const float* __restrict__ ba2, float* __restrict__ f1,
                       float* __restrict__ f2, int H, int Mr) {
  int i = blockIdx.x * blockDim.x + threadIdx.x;
  if (i >= H * Mr) return;
  int h = i / Mr;
  const float* s = sf + (long)i * 64;
  const float* A1 = a1 + h * 64;
  const float* A2 = a2 + h * 64;
  float v1 = 0.f, v2 = 0.f;
#pragma unroll 8
  for (int o = 0; o < 64; ++o) { float sv = s[o]; v1 += sv * A1[o]; v2 += sv * A2[o]; }
  f1[i] = v1 + ba1[h];
  f2[i] = v2 + ba2[h];
}

// sf [h,m,o] f32 -> sfT [h,o,m] f16 (B-operand for attention GEMM)
__global__ void k_sfT(const float* __restrict__ sf, _Float16* __restrict__ sfT,
                      int H, int Mr) {
  long i = (long)blockIdx.x * blockDim.x + threadIdx.x;
  long total = (long)H * 64 * Mr;
  if (i >= total) return;
  long t = i;
  int m = (int)(t % Mr); t /= Mr;
  int o = (int)(t % 64);
  int h = (int)(t / 64);
  sfT[i] = (_Float16)sf[((long)h * Mr + m) * 64 + o];
}

// pass 1: online softmax row stats per (h,b,i) row; one wave / row.
// Stores row max and RECIPROCAL of the exp-sum (epilogue becomes a multiply).
__global__ void k_rowstats(const float* __restrict__ f1, const float* __restrict__ f2,
                           const float* __restrict__ adj, float* __restrict__ rmax,
                           float* __restrict__ rinv, int H, int N, int B) {
  int wave = threadIdx.x >> 5, lane = threadIdx.x & 31;
  int gid = blockIdx.x * (blockDim.x >> 5) + wave;   // gid = h*B*N + b*N + i
  if (gid >= H * B * N) return;
  int i = gid % N;
  int hb = gid / N;
  int h = hb / B, b = hb % B;
  float f1v = f1[(long)h * B * N + (long)b * N + i];
  const float* f2h = f2 + (long)h * B * N + (long)b * N;
  const float* arow = adj + (long)i * N;
  float mx = -1e30f, sm = 0.f;
  for (int j = lane; j < N; j += 32) {
    float v = f1v + f2h[j];
    v = (v > 0.f) ? v : 0.2f * v;       // leaky_relu(0.2)
    v += arow[j];
    float nm = fmaxf(mx, v);
    sm = sm * __expf(mx - nm) + __expf(v - nm);
    mx = nm;
  }
#pragma unroll
  for (int off = 16; off >= 1; off >>= 1) {
    float om = __shfl_xor(mx, off, 32);
    float os = __shfl_xor(sm, off, 32);
    float nm = fmaxf(mx, om);
    sm = sm * __expf(mx - nm) + os * __expf(om - nm);
    mx = nm;
  }
  if (lane == 0) { rmax[gid] = mx; rinv[gid] = fast_rcp(sm); }
}

// pass 2: flash attention — regenerate P tiles in A-frag layout, WMMA vs sfT.
// A and B fragments double-buffered: next tile's exp/pack VALU and B loads
// overlap the current WMMA chain.
// attn[h,m,:] = (sum_j exp(lrelu(f1+f2)+adj - rmax) * sf[j,:]) * rinv
__global__ void k_attn(const float* __restrict__ f1, const float* __restrict__ f2,
                       const float* __restrict__ adj, const float* __restrict__ rmax,
                       const float* __restrict__ rinv, const _Float16* __restrict__ sfT,
                       float* __restrict__ attn, int H, int B, int N) {
  int wave = threadIdx.x >> 5, lane = threadIdx.x & 31;
  int Mr = B * N;
  int mbase = (blockIdx.x * 4 + wave) * 16;
  int h = blockIdx.y;
  int btile = mbase / N;                 // 16-row tile never straddles batches
  int m = mbase + (lane & 15);
  int ni = m - btile * N;
  float f1v = f1[(long)h * Mr + m];
  float rmx = rmax[(long)h * Mr + m];
  const float* f2h = f2 + (long)h * Mr + (long)btile * N;
  const float* arow = adj + (long)ni * N;
  const _Float16* Bth = sfT + (long)h * 64 * Mr + (long)btile * N + (lane & 15) * (long)Mr;

  v8f acc[4];
#pragma unroll
  for (int s = 0; s < 4; ++s)
#pragma unroll
    for (int r = 0; r < 8; ++r) acc[s][r] = 0.f;

  int kbA = (lane & 16) ? 16 : 0;
  int kbBo = (lane & 16) ? 16 : 0;

  // P-fragment generator for a 32-wide j tile (16 exps per lane, A-frag layout)
  auto make_p = [&](int j0) {
    v16h a;
#pragma unroll
    for (int g = 0; g < 2; ++g) {
      int jb = j0 + kbA + g * 8;
#pragma unroll
      for (int q = 0; q < 8; ++q) {
        int j = jb + q;
        float v = f1v + f2h[j];
        v = (v > 0.f) ? v : 0.2f * v;
        v += arow[j];
        a[g * 8 + q] = (_Float16)__expf(v - rmx);
      }
    }
    return a;
  };
  auto load_b = [&](int s, int j0) {
    return *(const v16h*)(Bth + (long)s * 16 * Mr + j0 + kbBo);
  };

  v16h a_cur = make_p(0);
  v16h b_cur[4];
#pragma unroll
  for (int s = 0; s < 4; ++s) b_cur[s] = load_b(s, 0);

  for (int j0 = 0; j0 < N; j0 += 32) {
    int jn = (j0 + 32 < N) ? (j0 + 32) : j0;          // branch-free tail
    v16h b_nxt[4];
#pragma unroll
    for (int s = 0; s < 4; ++s) b_nxt[s] = load_b(s, jn);
    v16h a_nxt = make_p(jn);
#pragma unroll
    for (int s = 0; s < 4; ++s)
      acc[s] = __builtin_amdgcn_wmma_f32_16x16x32_f16(false, a_cur, false, b_cur[s],
                                                      (short)0, acc[s], false, false);
    a_cur = a_nxt;
#pragma unroll
    for (int s = 0; s < 4; ++s) b_cur[s] = b_nxt[s];
  }
  int col = lane & 15, rofs = (lane & 16) ? 8 : 0;
#pragma unroll
  for (int r = 0; r < 8; ++r) {
    int row = mbase + r + rofs;
    float inv = rinv[(long)h * Mr + row];
#pragma unroll
    for (int s = 0; s < 4; ++s)
      attn[((long)h * Mr + row) * 64 + s * 16 + col] = acc[s][r] * inv;
  }
}

// block epilogue (concat heads): elu(attn+wr+br) [+res] -> relu -> bn(node) -> xnext [bn,H,64]
__global__ void k_epi_concat(const float* __restrict__ attn, const float* __restrict__ wr,
                             const float* __restrict__ br, const float* __restrict__ xprev,
                             const float* __restrict__ g, const float* __restrict__ be,
                             const float* __restrict__ mn, const float* __restrict__ vr,
                             float* __restrict__ xnext, int H, int B, int N,
                             int Tprev, int Cprev, int hasRes) {
  long i = (long)blockIdx.x * blockDim.x + threadIdx.x;
  long total = (long)B * N * H * 64;
  if (i >= total) return;
  int o = (int)(i & 63);
  long t = i >> 6;
  int h = (int)(t % H);
  long bn = t / H;
  int n = (int)(bn % N);
  long midx = ((long)h * B * N + bn) * 64 + o;
  float ret = attn[midx] + wr[midx] + br[h * 64 + o];
  float e = (ret > 0.f) ? ret : (__expf(ret) - 1.f);   // elu
  if (hasRes) e += xprev[(bn * Tprev + (Tprev - H + h)) * (long)Cprev + o];
  float rl = fmaxf(e, 0.f);
  xnext[i] = (rl - mn[n]) * fast_rsq(vr[n] + 1e-5f) * g[n] + be[n];
}

// block epilogue (mean heads, last block): mean(attn+wr+br) + res -> relu -> bn
__global__ void k_epi_mean(const float* __restrict__ attn, const float* __restrict__ wr,
                           const float* __restrict__ br, const float* __restrict__ xprev,
                           const float* __restrict__ g, const float* __restrict__ be,
                           const float* __restrict__ mn, const float* __restrict__ vr,
                           float* __restrict__ xnext, int H, int B, int N, int Tprev) {
  long i = (long)blockIdx.x * blockDim.x + threadIdx.x;
  long total = (long)B * N * 64;
  if (i >= total) return;
  int o = (int)(i & 63);
  long bn = i >> 6;
  int n = (int)(bn % N);
  float s = 0.f;
  for (int h = 0; h < H; ++h) {
    long midx = ((long)h * B * N + bn) * 64 + o;
    s += attn[midx] + wr[midx] + br[h * 64 + o];
  }
  s *= (1.f / (float)H);
  s += xprev[(bn * Tprev + (Tprev - 1)) * 64 + o];
  float rl = fmaxf(s, 0.f);
  xnext[i] = (rl - mn[n]) * fast_rsq(vr[n] + 1e-5f) * g[n] + be[n];
}

// fp32 [M,K] -> fp16 [M,Kpad] (zero pad)
__global__ void k_f32_to_f16(const float* __restrict__ in, _Float16* __restrict__ out,
                             int M, int K, int Kpad) {
  long i = (long)blockIdx.x * blockDim.x + threadIdx.x;
  long total = (long)M * Kpad;
  if (i >= total) return;
  int k = (int)(i % Kpad);
  long m = i / Kpad;
  out[i] = (k < K) ? (_Float16)in[m * K + k] : (_Float16)0.f;
}

// out-head epilogue 1: relu(raw+b1) -> bn(channel) -> f16 [4096,512]
__global__ void k_out1_epi(const float* __restrict__ raw, const float* __restrict__ b1,
                           const float* __restrict__ g, const float* __restrict__ be,
                           const float* __restrict__ mn, const float* __restrict__ vr,
                           _Float16* __restrict__ h16, int Mr) {
  long i = (long)blockIdx.x * blockDim.x + threadIdx.x;
  long total = (long)Mr * 512;
  if (i >= total) return;
  int e = (int)(i % 512);
  float v = fmaxf(raw[i] + b1[e], 0.f);
  v = (v - mn[e]) * fast_rsq(vr[e] + 1e-5f) * g[e] + be[e];
  h16[i] = (_Float16)v;
}

// W2 [12,512] -> Bt [16,512] fp16 (zero-padded rows)
__global__ void k_build_w2t(const float* __restrict__ W2, _Float16* __restrict__ out) {
  int i = blockIdx.x * blockDim.x + threadIdx.x;
  if (i >= 16 * 512) return;
  int e = i % 512, t = i / 512;
  out[i] = (t < 12) ? (_Float16)W2[t * 512 + e] : (_Float16)0.f;
}

// final: y = yraw[:, :12] + b2 -> d_out [B,N,12,1]
__global__ void k_out2_epi(const float* __restrict__ yraw, const float* __restrict__ b2,
                           float* __restrict__ out, int Mr) {
  int i = blockIdx.x * blockDim.x + threadIdx.x;
  if (i >= Mr * 12) return;
  int t = i % 12, m = i / 12;
  out[i] = yraw[m * 16 + t] + b2[t];
}

// ---------------------------------------------------------------------------
static inline unsigned cdiv(long a, long b) { return (unsigned)((a + b - 1) / b); }

extern "C" void kernel_launch(void* const* d_in, const int* in_sizes, int n_in,
                              void* d_out, int out_size, void* d_ws, size_t ws_size,
                              hipStream_t stream) {
  (void)in_sizes; (void)n_in; (void)out_size; (void)ws_size;
  const int B = 4, N = 1024, BN = 4096;

  // ---- input leaves (jax tree order: sorted dict keys, lists in order) ----
  int ii = 0;
  const float* A_hat = (const float*)d_in[ii++];
  const float* X     = (const float*)d_in[ii++];
  struct Blk { const float *W,*Wr,*a1,*a2,*ba1,*ba2,*bnb,*bng,*bnm,*bnv,*br,*tW1,*tW2,*tW3,*tb1,*tb2,*tb3; } blk[4];
  for (int b = 0; b < 4; ++b) {
    blk[b].W   = (const float*)d_in[ii++];   // 'W'
    blk[b].Wr  = (const float*)d_in[ii++];   // 'Wr'
    blk[b].a1  = (const float*)d_in[ii++];   // 'a1'
    blk[b].a2  = (const float*)d_in[ii++];   // 'a2'
    blk[b].ba1 = (const float*)d_in[ii++];   // 'b_a1'
    blk[b].ba2 = (const float*)d_in[ii++];   // 'b_a2'
    blk[b].bnb = (const float*)d_in[ii++];   // 'bn'.'beta'
    blk[b].bng = (const float*)d_in[ii++];   // 'bn'.'gamma'
    blk[b].bnm = (const float*)d_in[ii++];   // 'bn'.'mean'
    blk[b].bnv = (const float*)d_in[ii++];   // 'bn'.'var'
    blk[b].br  = (const float*)d_in[ii++];   // 'br'
    blk[b].tW1 = (const float*)d_in[ii++];   // 'tb'.'W1'
    blk[b].tW2 = (const float*)d_in[ii++];   // 'tb'.'W2'
    blk[b].tW3 = (const float*)d_in[ii++];   // 'tb'.'W3'
    blk[b].tb1 = (const float*)d_in[ii++];   // 'tb'.'b1'
    blk[b].tb2 = (const float*)d_in[ii++];   // 'tb'.'b2'
    blk[b].tb3 = (const float*)d_in[ii++];   // 'tb'.'b3'
  }
  const float* oW1  = (const float*)d_in[ii++];  // out.W1 [512,64]
  const float* oW2  = (const float*)d_in[ii++];  // out.W2 [12,512]
  const float* ob1  = (const float*)d_in[ii++];
  const float* ob2  = (const float*)d_in[ii++];
  const float* obnb = (const float*)d_in[ii++];
  const float* obng = (const float*)d_in[ii++];
  const float* obnm = (const float*)d_in[ii++];
  const float* obnv = (const float*)d_in[ii++];

  // ---- workspace carve (256B aligned) ----
  char* p = (char*)d_ws;
  auto alloc = [&](size_t bytes) { char* r = p; p += (bytes + 255) & ~(size_t)255; return r; };
  _Float16* col16  = (_Float16*)alloc(49152UL * 192 * 2);
  _Float16* wcat16 = (_Float16*)alloc(192UL * 192 * 2);
  float*    convraw= (float*)   alloc(49152UL * 192 * 4);
  _Float16* t16    = (_Float16*)alloc(4096UL * 768 * 2);
  _Float16* wt16   = (_Float16*)alloc(8UL * 64 * 768 * 2);
  _Float16* wrt16  = (_Float16*)alloc(8UL * 64 * 768 * 2);
  float*    sf     = (float*)   alloc(8UL * 4096 * 64 * 4);
  float*    wrout  = (float*)   alloc(8UL * 4096 * 64 * 4);
  _Float16* sfT16  = (_Float16*)alloc(8UL * 64 * 4096 * 2);
  float*    f1     = (float*)   alloc(8UL * 4096 * 4);
  float*    f2     = (float*)   alloc(8UL * 4096 * 4);
  float*    rmax   = (float*)   alloc(8UL * 4096 * 4);
  float*    rinv   = (float*)   alloc(8UL * 4096 * 4);
  float*    attn   = (float*)   alloc(8UL * 4096 * 64 * 4);
  float*    xA     = (float*)   alloc(4096UL * 6 * 64 * 4);
  float*    xB     = (float*)   alloc(4096UL * 6 * 64 * 4);
  _Float16* emb16  = (_Float16*)alloc(4096UL * 64 * 2);
  _Float16* w1t16  = (_Float16*)alloc(512UL * 64 * 2);
  float*    hraw   = (float*)   alloc(4096UL * 512 * 4);
  _Float16* h16    = (_Float16*)alloc(4096UL * 512 * 2);
  _Float16* w2t16  = (_Float16*)alloc(16UL * 512 * 2);
  float*    yraw   = (float*)   alloc(4096UL * 16 * 4);

  // ---- 4 STGAT blocks ----
  const float* xcur = X;             // [BN, 12, 2]
  float* xping[2] = {xA, xB};
  int Tcur = 12, Ccur = 2;
  for (int bl = 0; bl < 4; ++bl) {
    const Blk& q = blk[bl];
    int cin = Ccur, T = Tcur;
    int H = (bl == 3) ? 8 : 6;
    int inf = 64 * T;                        // 768 or 384 (K, mult of 32)
    int colKpad = (bl == 0) ? 32 : 192;      // 3*cin padded
    long Mconv = (long)BN * T;

    // fused 3-conv as one GEMM
    k_im2col<<<cdiv(Mconv * colKpad, 256), 256, 0, stream>>>(xcur, col16, BN, T, cin, colKpad);
    k_build_wcat<<<cdiv(192L * colKpad, 256), 256, 0, stream>>>(q.tW1, q.tW2, q.tW3, wcat16, cin, colKpad);
    {
      dim3 g((unsigned)(Mconv / 64), 3, 1);   // Nc=192 = 3 x 64
      k_gemm_t<4><<<g, 128, 0, stream>>>(col16, wcat16, convraw, (int)Mconv, colKpad, 192, 0, 0);
    }
    k_conv_epi<<<cdiv(Mconv * 64, 256), 256, 0, stream>>>(convraw, q.tb1, q.tb2, q.tb3, t16, Mconv);

    // per-head W / Wr -> transposed fp16
    {
      dim3 g(cdiv((long)inf * 64, 256), 1, (unsigned)H);
      k_transpose_bt<<<g, 256, 0, stream>>>(q.W,  wt16,  inf, (long)inf * 64, (long)64 * inf);
      k_transpose_bt<<<g, 256, 0, stream>>>(q.Wr, wrt16, inf, (long)inf * 64, (long)64 * inf);
    }
    // sf = t @ W[h], wrout = t @ Wr[h]   (batched over heads)
    {
      dim3 g((unsigned)(BN / 64), 1, (unsigned)H);   // Nc=64
      k_gemm_t<4><<<g, 128, 0, stream>>>(t16, wt16,  sf,    BN, inf, 64, (long)64 * inf, (long)BN * 64);
      k_gemm_t<4><<<g, 128, 0, stream>>>(t16, wrt16, wrout, BN, inf, 64, (long)64 * inf, (long)BN * 64);
    }
    k_f1f2<<<cdiv((long)H * BN, 256), 256, 0, stream>>>(sf, q.a1, q.a2, q.ba1, q.ba2, f1, f2, H, BN);
    k_sfT<<<cdiv((long)H * 64 * BN, 256), 256, 0, stream>>>(sf, sfT16, H, BN);

    // flash-style masked softmax-attention
    k_rowstats<<<cdiv((long)H * BN, 4), 128, 0, stream>>>(f1, f2, A_hat, rmax, rinv, H, N, B);
    {
      dim3 g((unsigned)(BN / 64), (unsigned)H, 1);
      k_attn<<<g, 128, 0, stream>>>(f1, f2, A_hat, rmax, rinv, sfT16, attn, H, B, N);
    }

    float* xnext = xping[bl & 1];
    if (bl < 3) {
      k_epi_concat<<<cdiv((long)BN * H * 64, 256), 256, 0, stream>>>(
          attn, wrout, q.br, xcur, q.bng, q.bnb, q.bnm, q.bnv, xnext,
          H, B, N, T, cin, (bl > 0) ? 1 : 0);
      Tcur = H; Ccur = 64;
    } else {
      k_epi_mean<<<cdiv((long)BN * 64, 256), 256, 0, stream>>>(
          attn, wrout, q.br, xcur, q.bng, q.bnb, q.bnm, q.bnv, xnext, H, B, N, T);
      Tcur = 1; Ccur = 64;
    }
    xcur = xnext;
  }

  // ---- output head ----
  k_f32_to_f16<<<cdiv((long)BN * 64, 256), 256, 0, stream>>>(xcur, emb16, BN, 64, 64);
  k_f32_to_f16<<<cdiv(512L * 64, 256), 256, 0, stream>>>(oW1, w1t16, 512, 64, 64);
  {
    dim3 g((unsigned)(BN / 64), 8, 1);   // Nc=512 = 8 x 64
    k_gemm_t<4><<<g, 128, 0, stream>>>(emb16, w1t16, hraw, BN, 64, 512, 0, 0);
  }
  k_out1_epi<<<cdiv((long)BN * 512, 256), 256, 0, stream>>>(hraw, ob1, obng, obnb, obnm, obnv, h16, BN);
  k_build_w2t<<<cdiv(16L * 512, 256), 256, 0, stream>>>(oW2, w2t16);
  {
    dim3 g((unsigned)(BN / 64), 1, 1);   // Nc=16 -> single 16-wide subtile
    k_gemm_t<1><<<g, 128, 0, stream>>>(h16, w2t16, yraw, BN, 512, 16, 0, 0);
  }
  k_out2_epi<<<cdiv((long)BN * 12, 256), 256, 0, stream>>>(yraw, ob2, (float*)d_out, BN);
}
